// ScaledDotProductAttention_19095424598671
// MI455X (gfx1250) — matrix operand order repaired and checked by reading, not compiled, not run on hardware
//
#include <hip/hip_runtime.h>

#define HIDDEN 1024
#define BATCH  8
#define L_Q    2048
#define L_K    2048

typedef __attribute__((ext_vector_type(16))) __bf16 bf16x16;
typedef __attribute__((ext_vector_type(8)))  __bf16 bf16x8;
typedef __attribute__((ext_vector_type(4)))  __bf16 bf16x4;
typedef __attribute__((ext_vector_type(8)))  float  f32x8;

union ABFrag { bf16x16 v; bf16x8 h[2]; };

static constexpr int QS_PITCH = HIDDEN + 8;   // bf16 elems (pitch = 2064 B, 16B mult)
static constexpr int SC_PITCH = L_K + 16;     // f32 elems  (pitch = 8256 B, 16B mult)
static constexpr int PB_PITCH = L_K + 8;      // bf16 elems (pitch = 4112 B, 16B mult)

// ---------------------------------------------------------------------------
// Kernel 1: convert K (f32) -> bf16, and also produce bf16 K^T via LDS tiles.
// Grid: (HIDDEN/32, L_K/32, BATCH), 256 threads.
// ---------------------------------------------------------------------------
__global__ void __launch_bounds__(256)
convert_transpose_k(const float* __restrict__ K,
                    unsigned short* __restrict__ Kbf_us,
                    unsigned short* __restrict__ KT_us)
{
    __bf16* Kbf = reinterpret_cast<__bf16*>(Kbf_us);
    __bf16* KT  = reinterpret_cast<__bf16*>(KT_us);
    __shared__ __bf16 tile[32][33];

    const int b  = blockIdx.z;
    const int k0 = blockIdx.y * 32;
    const int h0 = blockIdx.x * 32;
    const int tx = threadIdx.x & 31;
    const int tg = threadIdx.x >> 5;   // 8 row groups

    const float* Kb = K + (size_t)b * L_K * HIDDEN;
    #pragma unroll
    for (int r = tg; r < 32; r += 8) {
        float v = Kb[(size_t)(k0 + r) * HIDDEN + h0 + tx];
        __bf16 bv = (__bf16)v;
        Kbf[(size_t)b * L_K * HIDDEN + (size_t)(k0 + r) * HIDDEN + h0 + tx] = bv;
        tile[r][tx] = bv;
    }
    __syncthreads();
    #pragma unroll
    for (int r = tg; r < 32; r += 8) {
        KT[(size_t)b * HIDDEN * L_K + (size_t)(h0 + r) * L_K + k0 + tx] = tile[tx][r];
    }
}

// ---------------------------------------------------------------------------
// Kernel 2: fused attention. One workgroup per (batch, 16-row Q tile).
// 256 threads = 8 wave32s.
// LDS: Q tile (bf16) + score strip (f32) + probability strip (bf16) ~= 231 KB.
// ---------------------------------------------------------------------------
__global__ void __launch_bounds__(256)
attn_fused(const float* __restrict__ Q,
           const unsigned short* __restrict__ Kbf_us,
           const unsigned short* __restrict__ KT_us,
           const float* __restrict__ Mask,
           float* __restrict__ Ctx,
           float* __restrict__ Attn)
{
    extern __shared__ char smem[];
    __bf16* qs = reinterpret_cast<__bf16*>(smem);
    float*  sc = reinterpret_cast<float*>(smem + (size_t)16 * QS_PITCH * sizeof(__bf16));
    __bf16* pb = reinterpret_cast<__bf16*>(smem + (size_t)16 * QS_PITCH * sizeof(__bf16)
                                                + (size_t)16 * SC_PITCH * sizeof(float));

    const __bf16* Kbf = reinterpret_cast<const __bf16*>(Kbf_us);
    const __bf16* KT  = reinterpret_cast<const __bf16*>(KT_us);

    const int q0   = blockIdx.x * 16;
    const int b    = blockIdx.y;
    const int tid  = threadIdx.x;
    const int lane = tid & 31;
    const int wave = tid >> 5;
    const int nrow = lane & 15;   // N (or M for A frag) index within tile
    const int hf   = lane >> 4;   // lane-half selector

    // ---------------- Phase A: stage Q tile into LDS as bf16 ----------------
    const float* Qb = Q + ((size_t)b * L_Q + q0) * HIDDEN;
    for (int i = tid; i < 16 * HIDDEN / 4; i += 256) {
        int flat = i * 4;
        int r = flat >> 10;            // / HIDDEN
        int c = flat & (HIDDEN - 1);
        float4 v = *reinterpret_cast<const float4*>(Qb + (size_t)r * HIDDEN + c);
        __bf16* dst = qs + r * QS_PITCH + c;
        dst[0] = (__bf16)v.x; dst[1] = (__bf16)v.y;
        dst[2] = (__bf16)v.z; dst[3] = (__bf16)v.w;
    }
    __syncthreads();

    // -------- Phase B: S = (Q K^T)/32 * mask -> LDS (2 col-tiles/iter) --------
    const float scale = 0.03125f;  // 1/sqrt(1024)
    #pragma unroll 1
    for (int j = 0; j < 8; ++j) {
        const int kc0 = (wave + j * 8) * 32;   // pair of 16-wide tiles
        f32x8 acc0 = {}, acc1 = {};
        const __bf16* Bb0 = Kbf + ((size_t)b * L_K + kc0 + nrow) * HIDDEN + hf * 16;
        const __bf16* Bb1 = Bb0 + (size_t)16 * HIDDEN;
        const __bf16* Ab  = qs + nrow * QS_PITCH + hf * 16;
        #pragma unroll 4
        for (int kk = 0; kk < HIDDEN / 32; ++kk) {
            ABFrag a, b0, b1;
            a.h[0]  = *reinterpret_cast<const bf16x8*>(Ab + kk * 32);
            a.h[1]  = *reinterpret_cast<const bf16x8*>(Ab + kk * 32 + 8);
            b0.h[0] = *reinterpret_cast<const bf16x8*>(Bb0 + kk * 32);
            b0.h[1] = *reinterpret_cast<const bf16x8*>(Bb0 + kk * 32 + 8);
            b1.h[0] = *reinterpret_cast<const bf16x8*>(Bb1 + kk * 32);
            b1.h[1] = *reinterpret_cast<const bf16x8*>(Bb1 + kk * 32 + 8);
            acc0 = __builtin_amdgcn_wmma_f32_16x16x32_bf16(
                false, a.v, false, b0.v, (short)0, acc0, false, false);
            acc1 = __builtin_amdgcn_wmma_f32_16x16x32_bf16(
                false, a.v, false, b1.v, (short)0, acc1, false, false);
        }
        // C layout: lane (16*hf + n), VGPR v -> (M = v + 8*hf, N = n)
        const float* Mb = Mask + ((size_t)b * L_Q + q0) * L_K + kc0 + nrow;
        #pragma unroll
        for (int v = 0; v < 8; ++v) {
            int m = v + hf * 8;
            float mk0 = Mb[(size_t)m * L_K];
            float mk1 = Mb[(size_t)m * L_K + 16];
            sc[m * SC_PITCH + kc0 + nrow]      = acc0[v] * scale * mk0;
            sc[m * SC_PITCH + kc0 + 16 + nrow] = acc1[v] * scale * mk1;
        }
    }
    __syncthreads();

    // ------- Phase C: row softmax; write attn (f32) + P (bf16 in LDS) -------
    #pragma unroll
    for (int r = wave * 2; r < wave * 2 + 2; ++r) {
        float* row = sc + r * SC_PITCH;
        float mx = -3.402823466e38f;
        for (int c = lane * 4; c < L_K; c += 128) {
            float4 v = *reinterpret_cast<const float4*>(row + c);
            mx = fmaxf(mx, fmaxf(fmaxf(v.x, v.y), fmaxf(v.z, v.w)));
        }
        #pragma unroll
        for (int off = 16; off > 0; off >>= 1) mx = fmaxf(mx, __shfl_xor(mx, off, 32));

        float sum = 0.0f;
        for (int c = lane * 4; c < L_K; c += 128) {
            float4 v = *reinterpret_cast<const float4*>(row + c);
            v.x = __expf(v.x - mx); v.y = __expf(v.y - mx);
            v.z = __expf(v.z - mx); v.w = __expf(v.w - mx);
            *reinterpret_cast<float4*>(row + c) = v;
            sum += v.x + v.y + v.z + v.w;
        }
        #pragma unroll
        for (int off = 16; off > 0; off >>= 1) sum += __shfl_xor(sum, off, 32);

        float inv = 1.0f / sum;
        float*  arow = Attn + ((size_t)b * L_Q + q0 + r) * L_K;
        __bf16* prow = pb + r * PB_PITCH;
        for (int c = lane * 4; c < L_K; c += 128) {
            float4 v = *reinterpret_cast<const float4*>(row + c);
            v.x *= inv; v.y *= inv; v.z *= inv; v.w *= inv;
            *reinterpret_cast<float4*>(arow + c) = v;
            bf16x4 p;
            p[0] = (__bf16)v.x; p[1] = (__bf16)v.y;
            p[2] = (__bf16)v.z; p[3] = (__bf16)v.w;
            *reinterpret_cast<bf16x4*>(prow + c) = p;
        }
    }
    __syncthreads();

    // -------- Phase D: C = P K (2 col-tiles/iter, bf16 P from LDS) --------
    #pragma unroll 1
    for (int j = 0; j < 4; ++j) {
        const int h0 = (wave + j * 8) * 32;    // pair of 16-wide tiles
        f32x8 acc0 = {}, acc1 = {};
        const __bf16* Bb0 = KT + ((size_t)b * HIDDEN + h0 + nrow) * L_K + hf * 16;
        const __bf16* Bb1 = Bb0 + (size_t)16 * L_K;
        const __bf16* Ab  = pb + nrow * PB_PITCH + hf * 16;
        #pragma unroll 4
        for (int kk = 0; kk < L_K / 32; ++kk) {
            ABFrag a, b0, b1;
            a.h[0]  = *reinterpret_cast<const bf16x8*>(Ab + kk * 32);
            a.h[1]  = *reinterpret_cast<const bf16x8*>(Ab + kk * 32 + 8);
            b0.h[0] = *reinterpret_cast<const bf16x8*>(Bb0 + kk * 32);
            b0.h[1] = *reinterpret_cast<const bf16x8*>(Bb0 + kk * 32 + 8);
            b1.h[0] = *reinterpret_cast<const bf16x8*>(Bb1 + kk * 32);
            b1.h[1] = *reinterpret_cast<const bf16x8*>(Bb1 + kk * 32 + 8);
            acc0 = __builtin_amdgcn_wmma_f32_16x16x32_bf16(
                false, a.v, false, b0.v, (short)0, acc0, false, false);
            acc1 = __builtin_amdgcn_wmma_f32_16x16x32_bf16(
                false, a.v, false, b1.v, (short)0, acc1, false, false);
        }
        float* Cb = Ctx + ((size_t)b * L_Q + q0) * HIDDEN + h0 + nrow;
        #pragma unroll
        for (int v = 0; v < 8; ++v) {
            Cb[(size_t)(v + hf * 8) * HIDDEN]      = acc0[v];
            Cb[(size_t)(v + hf * 8) * HIDDEN + 16] = acc1[v];
        }
    }
}

// ---------------------------------------------------------------------------
extern "C" void kernel_launch(void* const* d_in, const int* in_sizes, int n_in,
                              void* d_out, int out_size, void* d_ws, size_t ws_size,
                              hipStream_t stream)
{
    (void)in_sizes; (void)n_in; (void)out_size; (void)ws_size;

    const float* Q    = (const float*)d_in[0];
    const float* K    = (const float*)d_in[1];
    const float* Mask = (const float*)d_in[2];

    float* Ctx  = (float*)d_out;                          // [B, LQ, H]
    float* Attn = Ctx + (size_t)BATCH * L_Q * HIDDEN;     // [B, LQ, LK]

    unsigned short* Kbf = (unsigned short*)d_ws;                       // bf16 [B, LK, H]
    unsigned short* KT  = Kbf + (size_t)BATCH * L_K * HIDDEN;          // bf16 [B, H, LK]

    dim3 gK(HIDDEN / 32, L_K / 32, BATCH);
    convert_transpose_k<<<gK, 256, 0, stream>>>(K, Kbf, KT);

    size_t smem = (size_t)16 * QS_PITCH * sizeof(__bf16)   // Q tile (bf16)
                + (size_t)16 * SC_PITCH * sizeof(float)    // score strip (f32)
                + (size_t)16 * PB_PITCH * sizeof(__bf16);  // P strip (bf16)
    dim3 gA(L_Q / 16, BATCH);
    attn_fused<<<gA, 256, smem, stream>>>(Q, Kbf, KT, Mask, Ctx, Attn);
}
